// Indexer_42262478193004
// MI455X (gfx1250) — compile-verified, matrix operand order checked by reading, not run on hardware
//
#include <hip/hip_runtime.h>
#include <stdint.h>

#define T_TOK     4096
#define DIM       7168
#define N_HEADS   64
#define HEAD_DIM  128
#define ROPE_DIM  64
#define Q_LORA    1536
#define TOPK_K    2048
#define NUM_SLOTS 8192
#define QDIM      (N_HEADS * HEAD_DIM)   // 8192
#define LN_EPS    1e-6f
#define SM_SCALE  0.08838834764831845f   // 1/sqrt(128)

typedef __attribute__((ext_vector_type(16))) __bf16        v16bf;
typedef __attribute__((ext_vector_type(8)))  float         v8f;
typedef __attribute__((ext_vector_type(8)))  unsigned int  v8u;

// ---- helpers -------------------------------------------------------------

__device__ __forceinline__ unsigned short f2bf(float f) {
  unsigned int u = __builtin_bit_cast(unsigned int, f);
  unsigned int r = u + 0x7FFFu + ((u >> 16) & 1u);   // round-to-nearest-even
  return (unsigned short)(r >> 16);
}

__device__ __forceinline__ v16bf load_bf16x16(const unsigned short* p) {
  v8u r = *(const v8u*)p;                            // 32B -> 2x b128
  return __builtin_bit_cast(v16bf, r);
}

// CDNA5 async copy: global -> LDS, 16B per lane, tracked by ASYNCcnt.
__device__ __forceinline__ void async_load_b128_to_lds(unsigned lds_off, const void* gp) {
  unsigned long long ga = (unsigned long long)(uintptr_t)gp;
  asm volatile("global_load_async_to_lds_b128 %0, %1, off"
               :: "v"(lds_off), "v"(ga) : "memory");
}
__device__ __forceinline__ void wait_async0() {
  asm volatile("s_wait_asynccnt 0" ::: "memory");
}

// ---- kernel: zero kv output region --------------------------------------

__global__ void k_zero_kv(float* __restrict__ kv) {
  int i = blockIdx.x * 256 + threadIdx.x;
  if (i < NUM_SLOTS * HEAD_DIM) kv[i] = 0.f;
}

// ---- kernel: k = rope(layernorm(x @ wk)) -> kv cache + bf16 copy --------

__global__ void k_compute_k(const float* __restrict__ x, const float* __restrict__ wk,
                            const float* __restrict__ cosp, const float* __restrict__ sinp,
                            const float* __restrict__ knw, const float* __restrict__ knb,
                            const int* __restrict__ slot_mapping,
                            float* __restrict__ kv_out, unsigned short* __restrict__ kb) {
  const int t   = blockIdx.x;
  const int tid = threadIdx.x;                       // 128 threads
  __shared__ float xs[128];
  __shared__ float red[128];
  __shared__ float kn[128];

  float acc = 0.f;
  for (int i0 = 0; i0 < DIM; i0 += 128) {
    __syncthreads();
    xs[tid] = x[(size_t)t * DIM + i0 + tid];
    __syncthreads();
    #pragma unroll 8
    for (int i = 0; i < 128; ++i)
      acc += xs[i] * wk[(size_t)(i0 + i) * HEAD_DIM + tid];
  }

  red[tid] = acc; __syncthreads();
  for (int st = 64; st > 0; st >>= 1) { if (tid < st) red[tid] += red[tid + st]; __syncthreads(); }
  float mu = red[0] * (1.f / 128.f);
  __syncthreads();
  float d = acc - mu;
  red[tid] = d * d; __syncthreads();
  for (int st = 64; st > 0; st >>= 1) { if (tid < st) red[tid] += red[tid + st]; __syncthreads(); }
  float var = red[0] * (1.f / 128.f);

  float v = d * rsqrtf(var + LN_EPS) * knw[tid] + knb[tid];
  kn[tid] = v;
  __syncthreads();

  float out = v;
  if (tid < ROPE_DIM) {
    float cv = cosp[(size_t)t * ROPE_DIM + tid];
    float sv = sinp[(size_t)t * ROPE_DIM + tid];
    if ((tid & 1) == 0) out = kn[tid] * cv - kn[tid + 1] * sv;   // r1 = x1*c - x2*s
    else                out = kn[tid] * cv + kn[tid - 1] * sv;   // r2 = x2*c + x1*s
  }
  int slot = slot_mapping[t];
  kv_out[(size_t)slot * HEAD_DIM + tid] = out;
  kb[(size_t)t * HEAD_DIM + tid] = f2bf(out);
}

// ---- kernel: weights = x @ w_weights ------------------------------------

__global__ void k_compute_w(const float* __restrict__ x, const float* __restrict__ ww,
                            float* __restrict__ wts) {
  const int t = blockIdx.x, tid = threadIdx.x;       // 64 threads
  __shared__ float xs[64];
  float acc = 0.f;
  for (int i0 = 0; i0 < DIM; i0 += 64) {
    __syncthreads();
    xs[tid] = x[(size_t)t * DIM + i0 + tid];
    __syncthreads();
    #pragma unroll 8
    for (int i = 0; i < 64; ++i)
      acc += xs[i] * ww[(size_t)(i0 + i) * N_HEADS + tid];
  }
  wts[(size_t)t * N_HEADS + tid] = acc;
}

// ---- kernel: qr (f32) -> qrb (bf16), elementwise ------------------------

__global__ void k_cvt_qr(const float* __restrict__ qr, unsigned short* __restrict__ qrb) {
  size_t i = (size_t)blockIdx.x * 256 + threadIdx.x;
  qrb[i] = f2bf(qr[i]);
}

// ---- kernel: wq_b (K x N, f32) -> wqt (N x K, bf16), 16x16 LDS tiles ----

__global__ void k_transpose_wqb(const float* __restrict__ wq_b,
                                unsigned short* __restrict__ wqt) {
  __shared__ unsigned short tile[16][17];            // +1 to dodge bank conflicts
  const int k0 = blockIdx.x * 16;                    // K: 1536/16 = 96
  const int n0 = blockIdx.y * 16;                    // N: 8192/16 = 512
  const int tx = threadIdx.x & 15, ty = threadIdx.x >> 4;  // 256 threads
  tile[ty][tx] = f2bf(wq_b[(size_t)(k0 + ty) * QDIM + n0 + tx]);
  __syncthreads();
  wqt[(size_t)(n0 + ty) * Q_LORA + k0 + tx] = tile[tx][ty];
}

// ---- kernel: q = rope_head(qrb @ wqt^T), bf16 out (WMMA) ----------------
// block: 16 rows x 128 cols (one head), 8 waves x one 16x16 C tile.
// K staged 128-wide into LDS via async copy -> 4 WMMA per stage.

__global__ void k_compute_q(const unsigned short* __restrict__ qrb,
                            const unsigned short* __restrict__ wqt,
                            const float* __restrict__ cosp, const float* __restrict__ sinp,
                            unsigned short* __restrict__ qb) {
  const int nb   = blockIdx.x;                       // head 0..63
  const int m0   = blockIdx.y * 16;                  // token rows
  const int tid  = threadIdx.x;                      // 256
  const int wv   = tid >> 5;
  const int lane = tid & 31;
  const int n0   = nb * 128 + wv * 16;

  __shared__ __attribute__((aligned(32))) unsigned short A_sh[16 * 128];
  __shared__ float c_sh[8][16 * 16];

  const int arow = tid >> 4;                         // 0..15
  const int aseg = tid & 15;                         // 16B segment
  const unsigned lds_off =
      (unsigned)(uintptr_t)&A_sh[0] + (unsigned)(arow * 256 + aseg * 16);

  v8f cacc = {};
  for (int kk = 0; kk < Q_LORA; kk += 128) {
    __syncthreads();                                 // done reading prev stage
    async_load_b128_to_lds(lds_off, qrb + (size_t)(m0 + arow) * Q_LORA + kk + aseg * 8);
    wait_async0();
    __syncthreads();                                 // 16x128 bf16 A tile ready

    #pragma unroll
    for (int kki = 0; kki < 4; ++kki) {
      int kc = kki * 32 + (lane >> 4) * 16;
      v16bf a = load_bf16x16(&A_sh[(lane & 15) * 128 + kc]);
      v16bf b = load_bf16x16(wqt + (size_t)(n0 + (lane & 15)) * Q_LORA + kk + kc);
      cacc = __builtin_amdgcn_wmma_f32_16x16x32_bf16(false, a, false, b,
                                                     (short)0, cacc, false, false);
    }
  }

  #pragma unroll
  for (int r = 0; r < 8; ++r) {                      // C layout: VGPR r -> M=r / r+8
    int row = r + ((lane & 16) ? 8 : 0);
    c_sh[wv][row * 16 + (lane & 15)] = cacc[r];
  }
  __syncthreads();

  for (int i = tid; i < 16 * 128; i += 256) {        // interleaved RoPE + store
    int row  = i >> 7;
    int dcol = i & 127;
    float v = c_sh[dcol >> 4][row * 16 + (dcol & 15)];
    float out = v;
    if (dcol < ROPE_DIM) {
      int t = m0 + row;
      float cv = cosp[(size_t)t * ROPE_DIM + dcol];
      float sv = sinp[(size_t)t * ROPE_DIM + dcol];
      int dp = dcol ^ 1;
      float p = c_sh[dp >> 4][row * 16 + (dp & 15)];
      out = (dcol & 1) ? (v * cv + p * sv) : (v * cv - p * sv);
    }
    qb[(size_t)(m0 + row) * QDIM + nb * 128 + dcol] = f2bf(out);
  }
}

// ---- kernel: scores(t,s) = sum_h w(t,h)*relu(scale * q_h(t)·k(s)) -------
// block: 16 t-rows x 256 s-cols; 8 waves x TWO 16x16 tiles (A reuse);
// 64 heads x (4 K-chunks x 2 tiles) WMMA.

__global__ void k_scores(const unsigned short* __restrict__ qb,
                         const unsigned short* __restrict__ kb,
                         const float* __restrict__ wts,
                         float* __restrict__ scores) {
  const int sb   = blockIdx.x;                       // 256-wide s block
  const int t0   = blockIdx.y * 16;
  const int tid  = threadIdx.x;
  const int wv   = tid >> 5, lane = tid & 31;
  const int s0   = sb * 256 + wv * 32;               // this wave: [s0, s0+32)
  const int sbase = sb * 256;

  if (sbase > t0 + 15) {                             // fully above diagonal
    for (int i = tid; i < 16 * 256; i += 256) {
      int row = i >> 8, col = i & 255;
      scores[(size_t)(t0 + row) * T_TOK + sbase + col] = -__builtin_inff();
    }
    return;
  }

  __shared__ float w_sh[16 * 64];
  __shared__ __attribute__((aligned(32))) unsigned short a_sh[16 * 128];

  for (int i = tid; i < 16 * 64; i += 256) {
    int row = i >> 6, h = i & 63;
    w_sh[i] = wts[(size_t)(t0 + row) * N_HEADS + h];
  }

  float acc0[8], acc1[8];
  #pragma unroll
  for (int r = 0; r < 8; ++r) { acc0[r] = 0.f; acc1[r] = 0.f; }

  const int arow = tid >> 4;                         // 0..15
  const int aseg = tid & 15;                         // 16B segment
  const unsigned lds_off =
      (unsigned)(uintptr_t)&a_sh[0] + (unsigned)(arow * 256 + aseg * 16);

  for (int h = 0; h < N_HEADS; ++h) {
    __syncthreads();                                 // done reading prev tile
    async_load_b128_to_lds(lds_off, qb + (size_t)(t0 + arow) * QDIM + h * 128 + aseg * 8);
    if (h + 1 < N_HEADS)
      __builtin_prefetch(qb + (size_t)(t0 + arow) * QDIM + (h + 1) * 128 + aseg * 8, 0, 0);
    wait_async0();
    __syncthreads();                                 // 4KB q head-tile resident

    v8f qk0 = {}, qk1 = {};
    #pragma unroll
    for (int kki = 0; kki < 4; ++kki) {              // K = 128 in 4x32 chunks
      int kc = kki * 32 + (lane >> 4) * 16;
      v16bf a  = load_bf16x16(&a_sh[(lane & 15) * 128 + kc]);
      v16bf b0 = load_bf16x16(kb + (size_t)(s0      + (lane & 15)) * HEAD_DIM + kc);
      v16bf b1 = load_bf16x16(kb + (size_t)(s0 + 16 + (lane & 15)) * HEAD_DIM + kc);
      qk0 = __builtin_amdgcn_wmma_f32_16x16x32_bf16(false, a, false, b0,
                                                    (short)0, qk0, false, false);
      qk1 = __builtin_amdgcn_wmma_f32_16x16x32_bf16(false, a, false, b1,
                                                    (short)0, qk1, false, false);
    }
    #pragma unroll
    for (int r = 0; r < 8; ++r) {                    // relu + head weight
      int row = r + ((lane & 16) ? 8 : 0);
      float w = w_sh[row * 64 + h];
      float sc0 = qk0[r] * SM_SCALE; sc0 = sc0 > 0.f ? sc0 : 0.f;
      float sc1 = qk1[r] * SM_SCALE; sc1 = sc1 > 0.f ? sc1 : 0.f;
      acc0[r] += w * sc0;
      acc1[r] += w * sc1;
    }
  }

  #pragma unroll
  for (int r = 0; r < 8; ++r) {                      // causal mask + store
    int row = r + ((lane & 16) ? 8 : 0);
    int t = t0 + row;
    int s_a = s0 + (lane & 15);
    int s_b = s_a + 16;
    scores[(size_t)t * T_TOK + s_a] = (s_a <= t) ? acc0[r] : -__builtin_inff();
    scores[(size_t)t * T_TOK + s_b] = (s_b <= t) ? acc1[r] : -__builtin_inff();
  }
}

// ---- kernel: per-row top-2048 via radix-histogram threshold select ------

__global__ void k_topk(const float* __restrict__ scores, int* __restrict__ out) {
  const int t = blockIdx.x;
  const int tid = threadIdx.x;                       // 256
  __shared__ unsigned hist[256];
  __shared__ unsigned thr_bin, need_thr, cnt_sel, cnt_thr;

  hist[tid] = 0;
  if (tid == 0) { cnt_sel = 0; cnt_thr = 0; }
  __syncthreads();

  const float* row = scores + (size_t)t * T_TOK;
  unsigned ukeys[16];
  #pragma unroll
  for (int j = 0; j < 16; ++j) {
    int s = tid + j * 256;
    unsigned u = __builtin_bit_cast(unsigned, row[s]);
    u = (u & 0x80000000u) ? ~u : (u | 0x80000000u);  // order-preserving map
    ukeys[j] = u;
    atomicAdd(&hist[u >> 24], 1u);
  }
  __syncthreads();

  if (tid == 0) {
    unsigned cum = 0; int b;
    for (b = 255; b >= 0; --b) {
      if (cum + hist[b] >= (unsigned)TOPK_K) break;
      cum += hist[b];
    }
    thr_bin = (unsigned)b;
    need_thr = (unsigned)TOPK_K - cum;
  }
  __syncthreads();

  const unsigned tb = thr_bin, nd = need_thr;
  #pragma unroll
  for (int j = 0; j < 16; ++j) {
    int s = tid + j * 256;
    unsigned b = ukeys[j] >> 24;
    if (b > tb) {
      unsigned pos = atomicAdd(&cnt_sel, 1u);
      out[(size_t)t * TOPK_K + pos] = s;
    } else if (b == tb) {
      unsigned q = atomicAdd(&cnt_thr, 1u);
      if (q < nd) {
        unsigned pos = atomicAdd(&cnt_sel, 1u);
        out[(size_t)t * TOPK_K + pos] = s;
      }
    }
  }
}

// ---- launch --------------------------------------------------------------

extern "C" void kernel_launch(void* const* d_in, const int* in_sizes, int n_in,
                              void* d_out, int out_size, void* d_ws, size_t ws_size,
                              hipStream_t stream) {
  const float* x    = (const float*)d_in[0];
  const float* qr   = (const float*)d_in[1];
  const float* cosp = (const float*)d_in[2];
  const float* sinp = (const float*)d_in[3];
  // d_in[4] kv_cache_k (zeros) rebuilt on device
  const int*   slot = (const int*)d_in[5];
  // d_in[6] seq_len scalar (fixed T=4096)
  const float* wqb  = (const float*)d_in[7];
  const float* wk   = (const float*)d_in[8];
  const float* ww   = (const float*)d_in[9];
  const float* knw  = (const float*)d_in[10];
  const float* knb  = (const float*)d_in[11];

  int*   topk_out = (int*)d_out;
  float* kv_out   = (float*)d_out + (size_t)T_TOK * TOPK_K;

  char* ws = (char*)d_ws;
  unsigned short* qb  = (unsigned short*)(ws);                  // 64 MB bf16 q
  unsigned short* kb  = (unsigned short*)(ws + (67108864));     //  1 MB bf16 k
  float*          wts = (float*)(ws + (67108864 + 1048576));    //  1 MB f32 weights
  float*          sc  = (float*)(ws + (67108864 + 2097152));    // 64 MB f32 scores
  unsigned short* qrb = (unsigned short*)(ws + (136314880));    // 12 MB bf16 qr
  unsigned short* wqt = (unsigned short*)(ws + (136314880 + 12582912)); // 24 MB bf16 wq^T

  k_zero_kv      <<<(NUM_SLOTS * HEAD_DIM + 255) / 256, 256, 0, stream>>>(kv_out);
  k_compute_k    <<<T_TOK, 128, 0, stream>>>(x, wk, cosp, sinp, knw, knb, slot, kv_out, kb);
  k_compute_w    <<<T_TOK, 64, 0, stream>>>(x, ww, wts);
  k_cvt_qr       <<<(T_TOK * Q_LORA) / 256, 256, 0, stream>>>(qr, qrb);
  k_transpose_wqb<<<dim3(Q_LORA / 16, QDIM / 16), 256, 0, stream>>>(wqb, wqt);
  k_compute_q    <<<dim3(N_HEADS, T_TOK / 16), 256, 0, stream>>>(qrb, wqt, cosp, sinp, qb);
  k_scores       <<<dim3(T_TOK / 256, T_TOK / 16), 256, 0, stream>>>(qb, kb, wts, sc);
  k_topk         <<<T_TOK, 256, 0, stream>>>(sc, topk_out);
}
